// DenseLanguageGuidanceModule_9234179686663
// MI455X (gfx1250) — matrix operand order repaired and checked by reading, not compiled, not run on hardware
//
#include <hip/hip_runtime.h>
#include <hip/hip_bf16.h>

// ---------------------------------------------------------------------------
// CDNA5 (gfx1250) WMMA pipeline for the dense language-guidance module.
// Heavy matmuls: V_WMMA_F32_16X16X32_F16 (wave32), LDS staged via
// GLOBAL_LOAD_ASYNC_TO_LDS_B128 when operands are already f16.
// ---------------------------------------------------------------------------

typedef _Float16 f16;
typedef __attribute__((ext_vector_type(16))) _Float16 v16h;
typedef __attribute__((ext_vector_type(8)))  _Float16 v8h;
typedef __attribute__((ext_vector_type(8)))  float    v8f;

union Frag16 { v16h v; v8h h[2]; };

#define BM 64
#define BN 128
#define BK 32
#define LDSP 40   // f16 row stride in LDS (32 + 8 pad -> 80B, conflict-free b128)

template<typename T> struct is_f16        { static const bool value = false; };
template<>           struct is_f16<f16>   { static const bool value = true;  };

// low 32 bits of a generic shared-space pointer == wave-relative LDS offset
__device__ inline unsigned int lds_off(const void* p) {
  return (unsigned int)(unsigned long long)p;
}

// ---- staging helpers -------------------------------------------------------
// 16 f16 row, f16 source: 2x async DMA global->LDS (ASYNCcnt tracked)
__device__ inline void stage16(const f16* __restrict__ gp, f16* lp) {
  unsigned int       l = lds_off(lp);
  unsigned long long g = (unsigned long long)gp;
  asm volatile("global_load_async_to_lds_b128 %0, %1, off\n\t"
               "global_load_async_to_lds_b128 %0, %1, off offset:16"
               :: "v"(l), "v"(g) : "memory");
}
// 32 f16 row, f16 source: 4x async DMA
__device__ inline void stage32(const f16* __restrict__ gp, f16* lp) {
  unsigned int       l = lds_off(lp);
  unsigned long long g = (unsigned long long)gp;
  asm volatile("global_load_async_to_lds_b128 %0, %1, off\n\t"
               "global_load_async_to_lds_b128 %0, %1, off offset:16\n\t"
               "global_load_async_to_lds_b128 %0, %1, off offset:32\n\t"
               "global_load_async_to_lds_b128 %0, %1, off offset:48"
               :: "v"(l), "v"(g) : "memory");
}
// f32 source: load + v_cvt_pk_f16_f32 + ds_store (DMA cannot convert)
__device__ inline void cvt16(const float* __restrict__ p, f16* t) {
  const float4* q = (const float4*)p;
  float4 x0 = q[0], x1 = q[1], x2 = q[2], x3 = q[3];
  t[0]=(f16)x0.x; t[1]=(f16)x0.y; t[2]=(f16)x0.z; t[3]=(f16)x0.w;
  t[4]=(f16)x1.x; t[5]=(f16)x1.y; t[6]=(f16)x1.z; t[7]=(f16)x1.w;
  t[8]=(f16)x2.x; t[9]=(f16)x2.y; t[10]=(f16)x2.z; t[11]=(f16)x2.w;
  t[12]=(f16)x3.x; t[13]=(f16)x3.y; t[14]=(f16)x3.z; t[15]=(f16)x3.w;
}
__device__ inline void stage16(const float* __restrict__ gp, f16* lp) {
  f16 tmp[16];
  cvt16(gp, tmp);
  __builtin_prefetch(gp + BK, 0, 1);       // next K-tile -> global_prefetch_b8
  *(v8h*)lp       = *(v8h*)&tmp[0];
  *(v8h*)(lp + 8) = *(v8h*)&tmp[8];
}
__device__ inline void stage32(const float* __restrict__ gp, f16* lp) {
  f16 tmp[32];
  cvt16(gp, tmp);
  cvt16(gp + 16, tmp + 16);
  __builtin_prefetch(gp + BK, 0, 1);
  *(v8h*)lp        = *(v8h*)&tmp[0];
  *(v8h*)(lp + 8)  = *(v8h*)&tmp[8];
  *(v8h*)(lp + 16) = *(v8h*)&tmp[16];
  *(v8h*)(lp + 24) = *(v8h*)&tmp[24];
}
__device__ inline void zero16(f16* lp) {
  v8h z = {};
  *(v8h*)lp = z; *(v8h*)(lp + 8) = z;
}
__device__ inline void zero32(f16* lp) {
  v8h z = {};
  *(v8h*)lp = z; *(v8h*)(lp + 8) = z; *(v8h*)(lp + 16) = z; *(v8h*)(lp + 24) = z;
}

// ---------------------------------------------------------------------------
// Batched GEMM: C[b] = scale * (A[b] x B[b]) + bias, f32 accumulation.
//   Block tile 64(M) x 128(N), BK=32; 4 waves, each 32x64 (2x4 WMMA tiles).
//   A: [M,K] row-major (TA float -> converted to f16 while staging)
//   B_IS_NK=false: Bm is [K,N] row-major (weights) -> transposed scatter
//   B_IS_NK=true : Bm is [N,K] row-major (math-B^T) -> direct row DMA
//   LDS layout: As[m][k], Bs[n][k] (fragment-friendly for both operands)
//   Columns in [N, Nstore) are stored as zeros (pads fm/P2 for later K loops).
// ---------------------------------------------------------------------------
template<typename TA, typename TB, bool B_IS_NK>
__global__ __launch_bounds__(128)
void gemm_wmma(const TA* __restrict__ A, const TB* __restrict__ Bm,
               const float* __restrict__ bias,
               float* __restrict__ Cf, f16* __restrict__ Ch,
               int M, int N, int K, int Kreal,
               int lda, int ldb, int ldc, int Nstore,
               long long sA, long long sB, long long sC,
               float scale)
{
  __shared__ f16 As[BM][LDSP];
  __shared__ f16 Bs[BN][LDSP];

  const bool USE_ASYNC = is_f16<TA>::value || (B_IS_NK && is_f16<TB>::value);

  const int bz = blockIdx.z;
  A  += (long long)bz * sA;
  Bm += (long long)bz * sB;
  const long long cbase = (long long)bz * sC;

  const int m0   = blockIdx.y * BM;
  const int n0   = blockIdx.x * BN;
  const int tid  = threadIdx.x;
  const int lane = tid & 31;
  const int wave = tid >> 5;
  const int wm   = (wave >> 1) * 32;   // wave offset in block tile (M)
  const int wn   = (wave & 1) * 64;    // wave offset in block tile (N)
  const int hl   = lane >> 4;          // lane half (0: lanes 0-15, 1: 16-31)
  const int lr   = lane & 15;

  v8f acc[2][4] = {};

  for (int k0 = 0; k0 < K; k0 += BK) {
    __syncthreads();
    // ---- stage A tile (64 x 32): 128 threads x 16 elems ----
    {
      const int row = tid >> 1;
      const int cs  = (tid & 1) * 16;
      const int gm  = m0 + row;
      if (gm < M) stage16(A + (long long)gm * lda + (k0 + cs), &As[row][cs]);
      else        zero16(&As[row][cs]);
    }
    // ---- stage B tile into Bs[n][k] ----
    if (B_IS_NK) {
      const int gn = n0 + tid;           // one 32-f16 row per thread
      if (gn < N) stage32(Bm + (long long)gn * ldb + k0, &Bs[tid][0]);
      else        zero32(&Bs[tid][0]);
    } else {
      const int kk = tid >> 2;           // 0..31 (k in tile)
      const int ns = (tid & 3) * 32;     // n segment
      const int gk = k0 + kk;
      if (gk < Kreal) {
        const TB* p = Bm + (long long)gk * ldb + (n0 + ns);
        #pragma unroll
        for (int j = 0; j < 32; ++j) {
          const int gn = n0 + ns + j;
          Bs[ns + j][kk] = (gn < N) ? (f16)p[j] : (f16)0.f;
        }
      } else {
        #pragma unroll
        for (int j = 0; j < 32; ++j) Bs[ns + j][kk] = (f16)0.f;
      }
    }
    if (USE_ASYNC) asm volatile("s_wait_asynccnt 0" ::: "memory");
    __syncthreads();

    // ---- build fragments per ISA 16-bit operand layouts ----
    // A 16x32: lanes 0-15 (row M=lr) hold K {0..7,16..23}; lanes 16-31 hold
    //          K {8..15,24..31}.   B 32x16: lanes 0-15 hold K {0..7,16..23} of col
    //          N=lr; lanes 16-31 hold K {8..15,24..31}.
    Frag16 af[2], bf[4];
    #pragma unroll
    for (int mi = 0; mi < 2; ++mi) {
      const f16* ap = &As[wm + mi * 16 + lr][hl * 8];
      af[mi].h[0] = *(const v8h*)ap;
      af[mi].h[1] = *(const v8h*)(ap + 16);
    }
    #pragma unroll
    for (int ni = 0; ni < 4; ++ni) {
      const f16* bp = &Bs[wn + ni * 16 + lr][hl * 8];
      bf[ni].h[0] = *(const v8h*)bp;
      bf[ni].h[1] = *(const v8h*)(bp + 16);
    }
    #pragma unroll
    for (int mi = 0; mi < 2; ++mi)
      #pragma unroll
      for (int ni = 0; ni < 4; ++ni)
        acc[mi][ni] = __builtin_amdgcn_wmma_f32_16x16x32_f16(
            false, af[mi].v, false, bf[ni].v,
            (short)0, acc[mi][ni], false, false);
  }

  // ---- store: C VGPR g holds M = g + 8*laneHalf, N = lane%16 ----
  #pragma unroll
  for (int mi = 0; mi < 2; ++mi) {
    #pragma unroll
    for (int ni = 0; ni < 4; ++ni) {
      const int gn = n0 + wn + ni * 16 + lr;
      if (gn >= Nstore) continue;
      const float bv = (bias != nullptr && gn < N) ? bias[gn] : 0.f;
      #pragma unroll
      for (int g = 0; g < 8; ++g) {
        const int gm = m0 + wm + mi * 16 + hl * 8 + g;
        if (gm >= M) continue;
        const float v = (gn < N) ? (acc[mi][ni][g] * scale + bv) : 0.f;
        const long long idx = cbase + (long long)gm * ldc + gn;
        if (Cf) Cf[idx] = v;
        if (Ch) Ch[idx] = (f16)v;
      }
    }
  }
}

// ---------------------------------------------------------------------------
// Softmax over last axis (rows of a_raw, length 1024). One wave per row.
// ---------------------------------------------------------------------------
__global__ __launch_bounds__(32)
void softmax_rows(const float* __restrict__ a, f16* __restrict__ P,
                  int cols)
{
  const int row  = blockIdx.x;
  const int lane = threadIdx.x;
  const float* ap = a + (long long)row * cols;
  float mx = -1e30f;
  for (int v = lane; v < cols; v += 32) mx = fmaxf(mx, ap[v]);
  #pragma unroll
  for (int o = 16; o; o >>= 1) mx = fmaxf(mx, __shfl_xor(mx, o, 32));
  float sum = 0.f;
  for (int v = lane; v < cols; v += 32) sum += __expf(ap[v] - mx);
  #pragma unroll
  for (int o = 16; o; o >>= 1) sum += __shfl_xor(sum, o, 32);
  const float inv = 1.f / sum;
  f16* pp = P + (long long)row * cols;
  for (int v = lane; v < cols; v += 32) pp[v] = (f16)(__expf(ap[v] - mx) * inv);
}

// ---------------------------------------------------------------------------
// Softmax over the Nl axis of a_raw^T: P2[b,v,l] = softmax_l(a_raw[b,l,v]).
// One wave per (b,v). Output padded with zeros to Nlp columns.
// ---------------------------------------------------------------------------
__global__ __launch_bounds__(32)
void softmax_cols(const float* __restrict__ a, f16* __restrict__ P2,
                  int Nl, int Nv, int Nlp)
{
  const int v    = blockIdx.x;
  const int b    = blockIdx.y;
  const int lane = threadIdx.x;
  const float* ap = a + (long long)b * Nl * Nv + v;

  float x[3];
  float mx = -1e30f;
  #pragma unroll
  for (int i = 0; i < 3; ++i) {
    const int l = lane + 32 * i;
    x[i] = (l < Nl) ? ap[(long long)l * Nv] : -1e30f;
    mx = fmaxf(mx, x[i]);
  }
  #pragma unroll
  for (int o = 16; o; o >>= 1) mx = fmaxf(mx, __shfl_xor(mx, o, 32));
  float e[3];
  float sum = 0.f;
  #pragma unroll
  for (int i = 0; i < 3; ++i) {
    const int l = lane + 32 * i;
    e[i] = (l < Nl) ? __expf(x[i] - mx) : 0.f;
    sum += e[i];
  }
  #pragma unroll
  for (int o = 16; o; o >>= 1) sum += __shfl_xor(sum, o, 32);
  const float inv = 1.f / sum;
  f16* pp = P2 + ((long long)b * Nv + v) * Nlp;
  #pragma unroll
  for (int i = 0; i < 3; ++i) {
    const int l = lane + 32 * i;
    if (l < Nlp) pp[l] = (f16)(e[i] * inv);   // zeros in [Nl, Nlp)
  }
}

// ---------------------------------------------------------------------------
// Host-side launch sequence
// ---------------------------------------------------------------------------
extern "C" void kernel_launch(void* const* d_in, const int* in_sizes, int n_in,
                              void* d_out, int out_size, void* d_ws, size_t ws_size,
                              hipStream_t stream) {
  (void)in_sizes; (void)n_in; (void)out_size; (void)ws_size;

  const int B  = 32, Nv = 1024, Dv = 768, Nl = 77, Dl = 512, D = 512, O = 768;
  const int Nlp = 96;                        // Nl padded to multiple of BK
  const float scale = 0.04419417382415922f;  // 1/sqrt(512)

  const float* fv   = (const float*)d_in[0];
  const float* fl   = (const float*)d_in[1];
  const float* W_vk = (const float*)d_in[2];
  const float* b_vk = (const float*)d_in[3];
  const float* W_vv = (const float*)d_in[4];
  const float* b_vv = (const float*)d_in[5];
  const float* W_lk = (const float*)d_in[6];
  const float* b_lk = (const float*)d_in[7];
  const float* W_lv = (const float*)d_in[8];
  const float* b_lv = (const float*)d_in[9];
  const float* W_m  = (const float*)d_in[10];
  const float* b_m  = (const float*)d_in[11];
  float* out = (float*)d_out;

  // Workspace carve-up (256B aligned)
  char* w = (char*)d_ws;
  size_t off = 0;
  auto alloc = [&](size_t bytes) -> void* {
    void* p = w + off;
    off += (bytes + 255) & ~(size_t)255;
    return p;
  };
  f16*   fk_v16 = (f16*)  alloc((size_t)B * Nv * D   * 2);
  f16*   fv_v16 = (f16*)  alloc((size_t)B * Nv * D   * 2);
  f16*   fk_l16 = (f16*)  alloc((size_t)B * Nl * D   * 2);
  f16*   fv_l16 = (f16*)  alloc((size_t)B * Nl * D   * 2);
  float* a_raw  = (float*)alloc((size_t)B * Nl * Nv  * 4);
  f16*   P16    = (f16*)  alloc((size_t)B * Nl * Nv  * 2);
  f16*   P2_16  = (f16*)  alloc((size_t)B * Nv * Nlp * 2);
  f16*   fa_v16 = (f16*)  alloc((size_t)B * Nl * D   * 2);
  f16*   fa_l16 = (f16*)  alloc((size_t)B * Nv * D   * 2);
  f16*   fm16   = (f16*)  alloc((size_t)B * Nv * Nlp * 2);

  auto grid = [](int N_, int M_, int Z_) {
    return dim3((N_ + BN - 1) / BN, (M_ + BM - 1) / BM, Z_);
  };
  const dim3 blk(128);

  // 1-2) fk_v = fv @ W_vk + b_vk ; fv_v = fv @ W_vv + b_vv     [B,1024,512] f16
  gemm_wmma<float, float, false><<<grid(D, Nv, B), blk, 0, stream>>>(
      fv, W_vk, b_vk, nullptr, fk_v16, Nv, D, Dv, Dv, Dv, D, D, D,
      (long long)Nv * Dv, 0, (long long)Nv * D, 1.f);
  gemm_wmma<float, float, false><<<grid(D, Nv, B), blk, 0, stream>>>(
      fv, W_vv, b_vv, nullptr, fv_v16, Nv, D, Dv, Dv, Dv, D, D, D,
      (long long)Nv * Dv, 0, (long long)Nv * D, 1.f);

  // 3-4) fk_l = fl @ W_lk + b_lk ; fv_l = fl @ W_lv + b_lv     [B,77,512] f16
  gemm_wmma<float, float, false><<<grid(D, Nl, B), blk, 0, stream>>>(
      fl, W_lk, b_lk, nullptr, fk_l16, Nl, D, Dl, Dl, Dl, D, D, D,
      (long long)Nl * Dl, 0, (long long)Nl * D, 1.f);
  gemm_wmma<float, float, false><<<grid(D, Nl, B), blk, 0, stream>>>(
      fl, W_lv, b_lv, nullptr, fv_l16, Nl, D, Dl, Dl, Dl, D, D, D,
      (long long)Nl * Dl, 0, (long long)Nl * D, 1.f);

  // 5) a_raw = fk_l @ fk_v^T * scale                           [B,77,1024] f32
  gemm_wmma<f16, f16, true><<<grid(Nv, Nl, B), blk, 0, stream>>>(
      fk_l16, fk_v16, nullptr, a_raw, nullptr, Nl, Nv, D, D, D, D, Nv, Nv,
      (long long)Nl * D, (long long)Nv * D, (long long)Nl * Nv, scale);

  // 6) P = softmax over Nv                                     [B,77,1024] f16
  softmax_rows<<<dim3(B * Nl), dim3(32), 0, stream>>>(a_raw, P16, Nv);

  // 7) P2 = softmax over Nl of a_raw^T, zero-padded to Nlp     [B,1024,96] f16
  softmax_cols<<<dim3(Nv, B), dim3(32), 0, stream>>>(a_raw, P2_16, Nl, Nv, Nlp);

  // 8) fa_v = P @ fv_v                                         [B,77,512] f16
  gemm_wmma<f16, f16, false><<<grid(D, Nl, B), blk, 0, stream>>>(
      P16, fv_v16, nullptr, nullptr, fa_v16, Nl, D, Nv, Nv, Nv, D, D, D,
      (long long)Nl * Nv, (long long)Nv * D, (long long)Nl * D, 1.f);

  // 9) fa_l = P2 @ fv_l   (K padded 77->96, loader zero-fills) [B,1024,512] f16
  gemm_wmma<f16, f16, false><<<grid(D, Nv, B), blk, 0, stream>>>(
      P2_16, fv_l16, nullptr, nullptr, fa_l16, Nv, D, Nlp, Nl, Nlp, D, D, D,
      (long long)Nv * Nlp, (long long)Nl * D, (long long)Nv * D, 1.f);

  // 10) fm = fa_l @ fa_v^T, columns padded with zeros to Nlp   [B,1024,96] f16
  gemm_wmma<f16, f16, true><<<grid(Nlp, Nv, B), blk, 0, stream>>>(
      fa_l16, fa_v16, nullptr, nullptr, fm16, Nv, Nl, D, D, D, D, Nlp, Nlp,
      (long long)Nv * D, (long long)Nl * D, (long long)Nv * Nlp, 1.f);

  // 11) out = fm @ W_m + b_m                                   [B,1024,768] f32
  gemm_wmma<f16, float, false><<<grid(O, Nv, B), blk, 0, stream>>>(
      fm16, W_m, b_m, out, nullptr, Nv, O, Nlp, Nl, Nlp, O, O, O,
      (long long)Nv * Nlp, 0, (long long)Nv * O, 1.f);
}
